// VQVAE_QE_12128987644314
// MI455X (gfx1250) — matrix operand order repaired and checked by reading, not compiled, not run on hardware
//
#include <hip/hip_runtime.h>
#include <hip/hip_bf16.h>

typedef __bf16 bf16;
typedef __attribute__((ext_vector_type(16))) __bf16 v16bf;
typedef __attribute__((ext_vector_type(8)))  float  v8f;

#define TSEQ 4096
#define NB   16
#define BTOT (NB * TSEQ)   // 65536 rows

// ---------------------------------------------------------------------------
// WMMA fragment helpers (CDNA5 16x16x32 bf16 layouts per ISA 7.12.2)
// A (16x32): lanes 0-15 -> M=lane, K = {0..7,16..23}; lanes 16-31 -> +8 on K
// B (32x16): lanes 0-15 -> N=lane, K = {0..7,16..23}; lanes 16-31 -> +8 on K
// D (16x16): lanes 0-15 -> N=lane, M=r; lanes 16-31 -> N=lane-16, M=r+8
// ---------------------------------------------------------------------------
__device__ inline v16bf ld_afrag(const bf16* p) {
  // p = row_base + k0 + half*8
  v16bf a;
#pragma unroll
  for (int j = 0; j < 4; ++j) { a[2*j]   = p[2*j];      a[2*j+1]   = p[2*j+1]; }
#pragma unroll
  for (int j = 0; j < 4; ++j) { a[8+2*j] = p[16+2*j];   a[9+2*j]   = p[16+2*j+1]; }
  return a;
}
__device__ inline v16bf ld_bfrag(const bf16* p) {
  // p = col_base + k0 + half*8
  v16bf b;
#pragma unroll
  for (int j = 0; j < 4; ++j) { b[2*j]   = p[2*j];      b[2*j+1]   = p[2*j+1]; }
#pragma unroll
  for (int j = 0; j < 4; ++j) { b[8+2*j] = p[16+2*j];   b[9+2*j]   = p[16+2*j+1]; }
  return b;
}
__device__ inline v8f wmma_bf16(v16bf a, v16bf b, v8f c) {
  return __builtin_amdgcn_wmma_f32_16x16x32_bf16(false, a, false, b, (short)0, c, false, false);
}
__device__ inline float sigm(float v) { return 1.f / (1.f + __expf(-v)); }

// gfx1250 async DMA: copy 16B global -> LDS, tracked on ASYNCcnt
__device__ inline void async_b128_to_lds(void* lds_dst, const void* gsrc) {
  unsigned l = (unsigned)(uintptr_t)lds_dst;  // low 32 bits = LDS offset
  asm volatile("global_load_async_to_lds_b128 %0, %1, off"
               :: "v"(l), "v"(gsrc) : "memory");
}
__device__ inline void wait_async0() {
  asm volatile("s_wait_asynccnt 0" ::: "memory");
}

// ---------------------------------------------------------------------------
// Weight prep: fp32 [Mact,Kact] row-major -> bf16 [rows,Kpad], zero padded
// ---------------------------------------------------------------------------
__global__ void k_prep_w(bf16* __restrict__ dst, const float* __restrict__ src,
                         int Mact, int Kact, int Kpad, int rows) {
  int i = blockIdx.x * 256 + threadIdx.x;
  if (i >= rows * Kpad) return;
  int r = i / Kpad, c = i - r * Kpad;
  float v = (r < Mact && c < Kact) ? src[(size_t)r * Kact + c] : 0.f;
  dst[i] = (bf16)v;
}

__global__ void k_prep_cb(bf16* __restrict__ cbbf, float* __restrict__ cbnorm,
                          const float* __restrict__ cb) {
  int i = blockIdx.x * 256 + threadIdx.x;
  if (i >= 1024) return;
  float s = 0.f;
  for (int d = 0; d < 64; ++d) {
    float v = cb[i * 64 + d];
    s += v * v;
    cbbf[i * 64 + d] = (bf16)v;
  }
  cbnorm[i] = 0.5f * s;
}

// ---------------------------------------------------------------------------
// Input transpose + InstanceNorm: x[B,T,257] -> xt[B,257,T] normalized over T
// ---------------------------------------------------------------------------
__global__ __launch_bounds__(256) void k_inorm_in(const float* __restrict__ x,
                                                  float* __restrict__ xt) {
  __shared__ float s1[256], s2[256];
  int bc = blockIdx.x, b = bc / 257, c = bc - b * 257, tid = threadIdx.x;
  float sum = 0.f, sq = 0.f;
  for (int t = tid; t < TSEQ; t += 256) {
    float v = x[((size_t)b * TSEQ + t) * 257 + c];
    sum += v; sq += v * v;
  }
  s1[tid] = sum; s2[tid] = sq;
  for (int off = 128; off; off >>= 1) {
    __syncthreads();
    if (tid < off) { s1[tid] += s1[tid + off]; s2[tid] += s2[tid + off]; }
  }
  __syncthreads();
  float m = s1[0] * (1.f / TSEQ);
  float var = s2[0] * (1.f / TSEQ) - m * m;
  float inv = rsqrtf(var + 1e-5f);
  for (int t = tid; t < TSEQ; t += 256) {
    float v = x[((size_t)b * TSEQ + t) * 257 + c];
    xt[((size_t)b * 257 + c) * TSEQ + t] = (v - m) * inv;
  }
}

// InstanceNorm on [B,C,T] contiguous (activation already applied at conv store)
__global__ __launch_bounds__(256) void k_inorm(const float* __restrict__ src,
                                               float* __restrict__ dst) {
  __shared__ float s1[256], s2[256];
  size_t base = (size_t)blockIdx.x * TSEQ;
  int tid = threadIdx.x;
  float sum = 0.f, sq = 0.f;
  for (int t = tid; t < TSEQ; t += 256) {
    float v = src[base + t]; sum += v; sq += v * v;
  }
  s1[tid] = sum; s2[tid] = sq;
  for (int off = 128; off; off >>= 1) {
    __syncthreads();
    if (tid < off) { s1[tid] += s1[tid + off]; s2[tid] += s2[tid + off]; }
  }
  __syncthreads();
  float m = s1[0] * (1.f / TSEQ);
  float var = s2[0] * (1.f / TSEQ) - m * m;
  float inv = rsqrtf(var + 1e-5f);
  for (int t = tid; t < TSEQ; t += 256) dst[base + t] = (src[base + t] - m) * inv;
}

// ---------------------------------------------------------------------------
// Conv1d (k=7, same pad) as implicit GEMM via WMMA bf16.
// Block: 8 waves, covers one (batch, 16-wide t tile); window + K-chunk patch
// staged in LDS; each wave owns up to 3 Cout tiles. A-fragment global loads
// are issued before the patch build so L2 weight latency overlaps DS work.
// act: 0 none, 1 lrelu(0.3), 3 relu.  out_btc: 0 -> [B,C,T], 1 -> [B,T,C]
// ---------------------------------------------------------------------------
__global__ __launch_bounds__(256) void k_conv(
    const float* __restrict__ src, const float* __restrict__ src2,
    const bf16* __restrict__ wbf, const float* __restrict__ bias,
    float* __restrict__ dst, int Cin, int Cout, int Kact, int Kpad,
    int act, int out_btc) {
  __shared__ bf16 win[257 * 22];
  __shared__ bf16 patch[512];
  const int t0 = blockIdx.x << 4;
  const int b = blockIdx.y;
  const int tid = threadIdx.x;
  const int nwin = Cin * 22;
  for (int i = tid; i < nwin; i += 256) {
    int c = i / 22, j = i - c * 22;
    int t = t0 - 3 + j;
    float v = 0.f;
    if ((unsigned)t < (unsigned)TSEQ) {
      size_t a = ((size_t)b * Cin + c) * TSEQ + t;
      v = src[a];
      if (src2) v += src2[a];
    }
    win[i] = (bf16)v;
  }
  const int lane = tid & 31, wave = tid >> 5;
  const int half = lane >> 4, ln = lane & 15;
  const int nmt = (Cout + 15) >> 4;
  const int mt0 = wave, mt1 = wave + 8, mt2 = wave + 16;
  v8f acc0 = {0.f,0.f,0.f,0.f,0.f,0.f,0.f,0.f};
  v8f acc1 = acc0, acc2 = acc0;
  const int nch = Kpad >> 5;
  for (int ch = 0; ch < nch; ++ch) {
    const int k0 = ch << 5;
    __syncthreads();  // window ready (ch=0) / previous consume done (ch>0)
    // issue A-fragment (weight) loads first: overlap with patch build below
    v16bf af0, af1, af2;
    if (mt0 < nmt) {
      const bf16* wr = wbf + (size_t)((mt0 << 4) + ln) * Kpad + k0 + (half << 3);
      __builtin_prefetch(wr + Kpad, 0, 0);
      af0 = ld_afrag(wr);
    }
    if (mt1 < nmt) {
      const bf16* wr = wbf + (size_t)((mt1 << 4) + ln) * Kpad + k0 + (half << 3);
      af1 = ld_afrag(wr);
    }
    if (mt2 < nmt) {
      const bf16* wr = wbf + (size_t)((mt2 << 4) + ln) * Kpad + k0 + (half << 3);
      af2 = ld_afrag(wr);
    }
    for (int i = tid; i < 512; i += 256) {
      int n = i >> 5, kl = i & 31, kk = k0 + kl;
      float v = 0.f;
      if (kk < Kact) {
        int c = kk / 7, k = kk - c * 7;
        v = (float)win[c * 22 + n + k];
      }
      patch[(n << 5) + kl] = (bf16)v;  // [n][kk] row-major, pairs contiguous
    }
    __syncthreads();
    v16bf bfr = ld_bfrag(&patch[(ln << 5) + (half << 3)]);
    if (mt0 < nmt) acc0 = wmma_bf16(af0, bfr, acc0);
    if (mt1 < nmt) acc1 = wmma_bf16(af1, bfr, acc1);
    if (mt2 < nmt) acc2 = wmma_bf16(af2, bfr, acc2);
  }
  const int t = t0 + ln;
  auto store_tile = [&](int mt, v8f acc) {
    if (mt >= nmt) return;
#pragma unroll
    for (int r = 0; r < 8; ++r) {
      int o = (mt << 4) + r + (half << 3);
      if (o < Cout) {
        float v = acc[r] + bias[o];
        if (act == 1) v = (v > 0.f) ? v : 0.3f * v;
        else if (act == 3) v = fmaxf(v, 0.f);
        size_t di = out_btc ? (((size_t)b * TSEQ + t) * (size_t)Cout + o)
                            : (((size_t)b * Cout + o) * (size_t)TSEQ + t);
        dst[di] = v;
      }
    }
  };
  store_tile(mt0, acc0); store_tile(mt1, acc1); store_tile(mt2, acc2);
}

// ---------------------------------------------------------------------------
// Linear: Y[BT,N] = act(X[BT,Kd] @ W^T + b), W prepped bf16 [N,Kd].
// Block = 16-row tile; wave w owns output tile w. act: 0 none,1 lrelu,2 sigmoid
// ---------------------------------------------------------------------------
__global__ __launch_bounds__(256) void k_linear(
    const float* __restrict__ X, const bf16* __restrict__ Wbf,
    const float* __restrict__ bias, float* __restrict__ Y,
    int Kd, int N, int act) {
  __shared__ bf16 sx[16 * 128];
  size_t row0 = (size_t)blockIdx.x * 16;
  int tid = threadIdx.x;
  int cnt = 16 * Kd;
  for (int i = tid; i < cnt; i += 256) sx[i] = (bf16)X[row0 * Kd + i];
  __syncthreads();
  const int lane = tid & 31, wave = tid >> 5;
  const int half = lane >> 4, ln = lane & 15;
  if (wave < (N >> 4)) {
    v8f acc = {0.f,0.f,0.f,0.f,0.f,0.f,0.f,0.f};
    for (int ch = 0; ch < (Kd >> 5); ++ch) {
      int k0 = ch << 5;
      v16bf bfr = ld_bfrag(Wbf + (size_t)((wave << 4) + ln) * Kd + k0 + (half << 3));
      v16bf afr = ld_afrag(&sx[ln * Kd + k0 + (half << 3)]);
      acc = wmma_bf16(afr, bfr, acc);
    }
    int col = (wave << 4) + ln;
    float bb = bias[col];
#pragma unroll
    for (int r = 0; r < 8; ++r) {
      size_t row = row0 + r + (half << 3);
      float v = acc[r] + bb;
      if (act == 1) v = (v > 0.f) ? v : 0.3f * v;
      else if (act == 2) v = sigm(v);
      Y[row * N + col] = v;
    }
  }
}

// ---------------------------------------------------------------------------
// BiLSTM: 32 persistent blocks (batch x dir), weights DMA'd to LDS via
// gfx1250 async global->LDS (ASYNCcnt), 128 threads = 128 gates, serial in T.
// ---------------------------------------------------------------------------
__global__ __launch_bounds__(128) void k_lstm(
    const float* __restrict__ z,  // [B,64,T]
    const float* __restrict__ Wih_f, const float* __restrict__ Whh_f,
    const float* __restrict__ bih_f, const float* __restrict__ bhh_f,
    const float* __restrict__ Wih_b, const float* __restrict__ Whh_b,
    const float* __restrict__ bih_b, const float* __restrict__ bhh_b,
    float* __restrict__ hcat) {   // [B,T,64] (fwd 0..31, bwd 32..63)
  __shared__ __align__(16) float sWih[128 * 64];
  __shared__ __align__(16) float sWhh[128 * 32];
  __shared__ float sb[128];
  __shared__ float sx[64], sh[32], sc[32], sg[128];
  const int bb = blockIdx.x & 15;
  const int dir = blockIdx.x >> 4;
  const float* Wih = dir ? Wih_b : Wih_f;
  const float* Whh = dir ? Whh_b : Whh_f;
  const float* bih = dir ? bih_b : bih_f;
  const float* bhh = dir ? bhh_b : bhh_f;
  const int tid = threadIdx.x;
  // async DMA weight stage: 16B per lane per issue, no VGPR round-trip
  for (int i = tid * 4; i < 128 * 64; i += 128 * 4)
    async_b128_to_lds(&sWih[i], &Wih[i]);
  for (int i = tid * 4; i < 128 * 32; i += 128 * 4)
    async_b128_to_lds(&sWhh[i], &Whh[i]);
  sb[tid] = bih[tid] + bhh[tid];
  if (tid < 32) { sh[tid] = 0.f; sc[tid] = 0.f; }
  wait_async0();
  __syncthreads();
  for (int step = 0; step < TSEQ; ++step) {
    const int t = dir ? (TSEQ - 1 - step) : step;
    if (tid < 64) sx[tid] = z[((size_t)bb * 64 + tid) * TSEQ + t];
    __syncthreads();
    float acc = sb[tid];
    const float* wr = &sWih[tid * 64];
#pragma unroll 8
    for (int d = 0; d < 64; ++d) acc += sx[d] * wr[d];
    const float* hr = &sWhh[tid * 32];
#pragma unroll 8
    for (int d = 0; d < 32; ++d) acc += sh[d] * hr[d];
    sg[tid] = acc;
    __syncthreads();
    if (tid < 32) {
      float ig = sigm(sg[tid]);
      float fg = sigm(sg[32 + tid]);
      float gg = tanhf(sg[64 + tid]);
      float og = sigm(sg[96 + tid]);
      float c = fg * sc[tid] + ig * gg;
      float h = og * tanhf(c);
      sc[tid] = c; sh[tid] = h;
      hcat[((size_t)bb * TSEQ + t) * 64 + dir * 32 + tid] = h;
    }
    __syncthreads();
  }
}

// ---------------------------------------------------------------------------
// LayerNorm over 64 features, affine; out row stride/offset configurable
// ---------------------------------------------------------------------------
__global__ __launch_bounds__(64) void k_ln64(
    const float* __restrict__ in, float* __restrict__ out,
    const float* __restrict__ g, const float* __restrict__ bta,
    int ostride, int ooff) {
  __shared__ float s[64];
  size_t row = blockIdx.x;
  int tid = threadIdx.x;
  float v = in[row * 64 + tid];
  s[tid] = v;
  for (int off = 32; off; off >>= 1) { __syncthreads(); if (tid < off) s[tid] += s[tid + off]; }
  __syncthreads();
  float m = s[0] * (1.f / 64.f);
  __syncthreads();
  float d = v - m;
  s[tid] = d * d;
  for (int off = 32; off; off >>= 1) { __syncthreads(); if (tid < off) s[tid] += s[tid + off]; }
  __syncthreads();
  float inv = rsqrtf(s[0] * (1.f / 64.f) + 1e-5f);
  out[row * ostride + ooff + tid] = d * inv * g[tid] + bta[tid];
}

// zc[row][0..63] = z[b][d][t]  (lower half of concat)
__global__ void k_zcopy(const float* __restrict__ z, float* __restrict__ zc) {
  size_t i = (size_t)blockIdx.x * 256 + threadIdx.x;
  if (i >= (size_t)BTOT * 64) return;
  size_t row = i >> 6;
  int d = (int)(i & 63);
  int b = (int)(row >> 12), t = (int)(row & 4095);
  zc[row * 128 + d] = z[((size_t)b * 64 + d) * TSEQ + t];
}

__global__ void k_mul(float* __restrict__ x, const float* __restrict__ a, size_t n) {
  size_t i = (size_t)blockIdx.x * 256 + threadIdx.x;
  if (i < n) x[i] *= a[i];
}

// ---------------------------------------------------------------------------
// VQ: fused GEMM(zs x cb^T) + argmin(0.5|cb|^2 - dot) + gather + loss partial
// One wave per 16-row tile; 128 WMMAs/block over 64 codebook tiles.
// ---------------------------------------------------------------------------
__global__ __launch_bounds__(32) void k_vq(
    const float* __restrict__ zs, const bf16* __restrict__ cbbf,
    const float* __restrict__ cbnorm, const float* __restrict__ cbf,
    float* __restrict__ q /*[B,64,T]*/, float* __restrict__ idx_out,
    float* __restrict__ part) {
  __shared__ bf16 sz[16 * 64];
  __shared__ float sdist[256];
  __shared__ float lp[16];
  const int tid = threadIdx.x;
  const size_t row0 = (size_t)blockIdx.x * 16;
  for (int i = tid; i < 1024; i += 32) sz[i] = (bf16)zs[row0 * 64 + i];
  __syncthreads();
  const int half = tid >> 4, ln = tid & 15;
  v16bf a0 = ld_afrag(&sz[ln * 64 + 0  + (half << 3)]);
  v16bf a1 = ld_afrag(&sz[ln * 64 + 32 + (half << 3)]);
  float bestv = 3.4e38f;
  int besti = 0;
  for (int ct = 0; ct < 64; ++ct) {
    const int cw = (ct << 4) + ln;
    v8f acc = {0.f,0.f,0.f,0.f,0.f,0.f,0.f,0.f};
    v16bf b0 = ld_bfrag(cbbf + (size_t)cw * 64 + 0  + (half << 3));
    v16bf b1 = ld_bfrag(cbbf + (size_t)cw * 64 + 32 + (half << 3));
    acc = wmma_bf16(a0, b0, acc);
    acc = wmma_bf16(a1, b1, acc);
    const float nrm = cbnorm[cw];
#pragma unroll
    for (int r = 0; r < 8; ++r) sdist[(r + (half << 3)) * 16 + ln] = nrm - acc[r];
    __syncthreads();
    if (tid < 16) {
#pragma unroll
      for (int c = 0; c < 16; ++c) {
        float v = sdist[tid * 16 + c];
        if (v < bestv) { bestv = v; besti = (ct << 4) + c; }
      }
    }
    __syncthreads();
  }
  if (tid < 16) {
    size_t g = row0 + tid;
    int b = (int)(g >> 12), t = (int)(g & 4095);
    idx_out[g] = (float)besti;
    float ls = 0.f;
    for (int d = 0; d < 64; ++d) {
      float qv = cbf[besti * 64 + d];
      q[((size_t)b * 64 + d) * TSEQ + t] = qv;
      float df = zs[g * 64 + d] - qv;
      ls += df * df;
    }
    lp[tid] = ls;
  }
  __syncthreads();
  if (tid == 0) {
    float s = 0.f;
    for (int i = 0; i < 16; ++i) s += lp[i];
    part[blockIdx.x] = s;   // deterministic per-block partial, no atomics
  }
}

__global__ __launch_bounds__(256) void k_loss_final(const float* __restrict__ part,
                                                    float* __restrict__ out) {
  __shared__ float s[256];
  int tid = threadIdx.x;
  float a = 0.f;
  for (int i = tid; i < 4096; i += 256) a += part[i];
  s[tid] = a;
  for (int off = 128; off; off >>= 1) { __syncthreads(); if (tid < off) s[tid] += s[tid + off]; }
  __syncthreads();
  if (tid == 0) out[0] = s[0] * (1.f / ((float)BTOT * 64.f));
}

// ---------------------------------------------------------------------------
// Host launch
// ---------------------------------------------------------------------------
extern "C" void kernel_launch(void* const* d_in, const int* in_sizes, int n_in,
                              void* d_out, int out_size, void* d_ws, size_t ws_size,
                              hipStream_t stream) {
  (void)in_sizes; (void)n_in; (void)out_size; (void)ws_size;
  auto P = [&](int i) { return (const float*)d_in[i]; };
  // d_in order: 0 x, 1 stochastic, 2.. params in dict order (see reference)
  float* out = (float*)d_out;
  float* idx_out  = out + 16842752;           // [B,T,257] then idx [B,T]
  float* loss_out = out + 16842752 + 65536;   // then scalar vqloss

  float* ws = (float*)d_ws;
  size_t fo = 0;
  auto A = [&](size_t n) { float* p = ws + fo; fo += n; return p; };
  float* xt   = A(16ull * 257 * 4096);
  float* tmp  = A(16ull * 128 * 4096);
  float* e1   = A(16ull * 128 * 4096);
  float* e2   = A(16ull * 128 * 4096);
  float* e3   = A(16ull * 64 * 4096);
  float* e4   = A(16ull * 64 * 4096);
  float* e5   = A(16ull * 64 * 4096);
  float* zb   = A(16ull * 64 * 4096);    // z as [B,64,T]
  float* hcat = A((size_t)BTOT * 64);
  float* zc   = A((size_t)BTOT * 128);
  float* part = A(4096);
  float* cbn  = A(1024);
  bf16* wb = (bf16*)(ws + fo);
  size_t bo = 0;
  auto BA = [&](size_t n) { bf16* p = wb + bo; bo += n; return p; };
  bf16* w_e1 = BA(128ull * 1824);
  bf16* w_e2 = BA(128ull * 896);
  bf16* w_e3 = BA(64ull * 896);
  bf16* w_e4 = BA(64ull * 448);
  bf16* w_e5 = BA(64ull * 448);
  bf16* w_e6 = BA(64ull * 448);
  bf16* w_d1 = BA(64ull * 448);
  bf16* w_d2 = BA(64ull * 448);
  bf16* w_d3 = BA(64ull * 448);
  bf16* w_d4 = BA(128ull * 448);
  bf16* w_d5 = BA(128ull * 896);
  bf16* w_d6 = BA(272ull * 896);
  bf16* w_a1 = BA(128ull * 128);
  bf16* w_a2 = BA(128ull * 128);
  bf16* w_p1 = BA(64ull * 128);
  bf16* w_p2 = BA(64ull * 64);
  bf16* cbbf = BA(1024ull * 64);

  // slab reuse
  float* t1 = e1;  float* aat = e2;
  float* h1a = e3; float* h1 = e4; float* zsr = e3; float* zs = e5;
  float* q = zb;
  float* d1 = e3;  float* d2 = e4;  float* d3 = e5;  float* d4 = e1;  float* d5 = e2;

  auto gp = [&](size_t n) { return dim3((unsigned)((n + 255) / 256)); };
  // ---- weight prep ----
  k_prep_w<<<gp(128ull*1824),256,0,stream>>>(w_e1, P(2),  128, 1799, 1824, 128);
  k_prep_w<<<gp(128ull*896), 256,0,stream>>>(w_e2, P(4),  128, 896,  896,  128);
  k_prep_w<<<gp(64ull*896),  256,0,stream>>>(w_e3, P(6),  64,  896,  896,  64);
  k_prep_w<<<gp(64ull*448),  256,0,stream>>>(w_e4, P(8),  64,  448,  448,  64);
  k_prep_w<<<gp(64ull*448),  256,0,stream>>>(w_e5, P(10), 64,  448,  448,  64);
  k_prep_w<<<gp(64ull*448),  256,0,stream>>>(w_e6, P(12), 64,  448,  448,  64);
  k_prep_w<<<gp(64ull*448),  256,0,stream>>>(w_d1, P(37), 64,  448,  448,  64);
  k_prep_w<<<gp(64ull*448),  256,0,stream>>>(w_d2, P(39), 64,  448,  448,  64);
  k_prep_w<<<gp(64ull*448),  256,0,stream>>>(w_d3, P(41), 64,  448,  448,  64);
  k_prep_w<<<gp(128ull*448), 256,0,stream>>>(w_d4, P(43), 128, 448,  448,  128);
  k_prep_w<<<gp(128ull*896), 256,0,stream>>>(w_d5, P(45), 128, 896,  896,  128);
  k_prep_w<<<gp(272ull*896), 256,0,stream>>>(w_d6, P(47), 257, 896,  896,  272);
  k_prep_w<<<gp(128ull*128), 256,0,stream>>>(w_a1, P(24), 128, 128,  128,  128);
  k_prep_w<<<gp(128ull*128), 256,0,stream>>>(w_a2, P(26), 128, 128,  128,  128);
  k_prep_w<<<gp(64ull*128),  256,0,stream>>>(w_p1, P(28), 64,  128,  128,  64);
  k_prep_w<<<gp(64ull*64),   256,0,stream>>>(w_p2, P(32), 64,  64,   64,   64);
  k_prep_cb<<<4,256,0,stream>>>(cbbf, cbn, P(36));

  dim3 cg(TSEQ / 16, NB);
  // ---- encoder ----
  k_inorm_in<<<16*257,256,0,stream>>>(P(0), xt);
  k_conv<<<cg,256,0,stream>>>(xt, nullptr, w_e1, P(3),  tmp, 257, 128, 1799, 1824, 1, 0);
  k_inorm<<<16*128,256,0,stream>>>(tmp, e1);
  k_conv<<<cg,256,0,stream>>>(e1, nullptr, w_e2, P(5),  tmp, 128, 128, 896, 896, 1, 0);
  k_inorm<<<16*128,256,0,stream>>>(tmp, e2);
  k_conv<<<cg,256,0,stream>>>(e1, e2,      w_e3, P(7),  tmp, 128, 64,  896, 896, 1, 0);
  k_inorm<<<16*64,256,0,stream>>>(tmp, e3);
  k_conv<<<cg,256,0,stream>>>(e3, nullptr, w_e4, P(9),  tmp, 64,  64,  448, 448, 1, 0);
  k_inorm<<<16*64,256,0,stream>>>(tmp, e4);
  k_conv<<<cg,256,0,stream>>>(e3, e4,      w_e5, P(11), tmp, 64,  64,  448, 448, 1, 0);
  k_inorm<<<16*64,256,0,stream>>>(tmp, e5);
  k_conv<<<cg,256,0,stream>>>(e5, nullptr, w_e6, P(13), tmp, 64,  64,  448, 448, 0, 0);
  k_inorm<<<16*64,256,0,stream>>>(tmp, zb);

  // ---- BiLSTM + layernorm + concat ----
  k_lstm<<<32,128,0,stream>>>(zb, P(14), P(15), P(16), P(17),
                              P(18), P(19), P(20), P(21), hcat);
  k_zcopy<<<gp((size_t)BTOT*64),256,0,stream>>>(zb, zc);
  k_ln64<<<BTOT,64,0,stream>>>(hcat, zc, P(22), P(23), 128, 64);

  // ---- EnhancedFeatureReduction ----
  k_linear<<<BTOT/16,256,0,stream>>>(zc, w_a1, P(25), t1,  128, 128, 1);
  k_linear<<<BTOT/16,256,0,stream>>>(t1, w_a2, P(27), aat, 128, 128, 2);
  k_mul<<<gp((size_t)BTOT*128),256,0,stream>>>(zc, aat, (size_t)BTOT*128);
  k_linear<<<BTOT/16,256,0,stream>>>(zc, w_p1, P(29), h1a, 128, 64, 1);
  k_ln64<<<BTOT,64,0,stream>>>(h1a, h1, P(30), P(31), 64, 0);
  k_linear<<<BTOT/16,256,0,stream>>>(h1, w_p2, P(33), zsr, 64, 64, 0);
  k_ln64<<<BTOT,64,0,stream>>>(zsr, zs, P(34), P(35), 64, 0);

  // ---- VQ ----
  k_vq<<<BTOT/16,32,0,stream>>>(zs, cbbf, cbn, P(36), q, idx_out, part);
  k_loss_final<<<1,256,0,stream>>>(part, loss_out);

  // ---- decoder ----
  k_conv<<<cg,256,0,stream>>>(q,  nullptr, w_d1, P(38), d1, 64,  64,  448, 448, 1, 0);
  k_conv<<<cg,256,0,stream>>>(d1, nullptr, w_d2, P(40), d2, 64,  64,  448, 448, 1, 0);
  k_conv<<<cg,256,0,stream>>>(d2, nullptr, w_d3, P(42), d3, 64,  64,  448, 448, 1, 0);
  k_conv<<<cg,256,0,stream>>>(d3, d2,      w_d4, P(44), d4, 64,  128, 448, 448, 1, 0);
  k_conv<<<cg,256,0,stream>>>(d4, nullptr, w_d5, P(46), d5, 128, 128, 896, 896, 1, 0);
  k_conv<<<cg,256,0,stream>>>(d5, d4,      w_d6, P(48), out, 128, 257, 896, 896, 3, 1);
}
